// MAResUNet_87445534147269
// MI455X (gfx1250) — matrix operand order repaired and checked by reading, not compiled, not run on hardware
//
#include <hip/hip_runtime.h>
#include <hip/hip_bf16.h>

// ---------------------------------------------------------------------------
// MI455X (gfx1250, wave32) implementation.
//   1. f32->f16 converts (x, y, pointwise/3x3 weights)
//   2. qkv 1x1 conv        : WMMA GEMM  M=768 K=256  P=16384/batch  -> f16
//   3. local2 1x1 conv + BN: WMMA GEMM  M=256 K=256                -> f32
//   4. local1 3x3 conv + BN: implicit-GEMM WMMA K=2304, accum into local
//   5. window attention    : per (window,head) wave; WMMA qk^T, softmax, WMMA av
//   6. avg-pool(8x1)+(1x8)+local fuse (VALU)
//   7. depthwise 8x8 + BN  (VALU) -> f16
//   8. projection 1x1 conv : WMMA GEMM -> d_out (f32)
// GEMM: 256 threads = 8 waves; block tile 128(m) x 128(p); each wave owns a
// 32m x 64n strip -> 8 WMMA per K-step against 12 ds_load_b128.
// ---------------------------------------------------------------------------

typedef __attribute__((ext_vector_type(16))) _Float16 v16h;
typedef __attribute__((ext_vector_type(8)))  _Float16 v8h;
typedef __attribute__((ext_vector_type(8)))  float    v8f;

#define CB   8
#define CDIM 256
#define CH   128
#define CW   128
#define CP   (CH * CW)
#define CWS  8
#define CHEADS 16

__device__ __forceinline__ v16h pack16(v8h lo, v8h hi) {
  return __builtin_shufflevector(lo, hi, 0, 1, 2, 3, 4, 5, 6, 7,
                                          8, 9, 10, 11, 12, 13, 14, 15);
}

__device__ __forceinline__ v8f wmma_f16(v16h a, v16h b, v8f c) {
  return __builtin_amdgcn_wmma_f32_16x16x32_f16(
      false, a, false, b, (short)0, c, false, false);
}

// ---------------------------------------------------------------------------
__global__ void cvt_f16_kernel(const float* __restrict__ src,
                               _Float16* __restrict__ dst, int n) {
  int i = blockIdx.x * blockDim.x + threadIdx.x;
  if (i < n) dst[i] = (_Float16)src[i];
}

// ---------------------------------------------------------------------------
// 1x1-conv GEMM: Out[b,m,p] = sum_k Wt[m,k] * X[b,k,p]  (+BN fold / accum)
// ---------------------------------------------------------------------------
template <int FUSE_BN, int ACCUM, int OUT_F16>
__global__ __launch_bounds__(256) void wmma_gemm_kernel(
    const _Float16* __restrict__ Wt,   // [M][K] f16
    const _Float16* __restrict__ X,    // [B][K][P] f16 channel-major
    float* __restrict__ OutF,
    _Float16* __restrict__ OutH,
    const float* __restrict__ bn_g, const float* __restrict__ bn_b,
    const float* __restrict__ bn_m, const float* __restrict__ bn_v,
    int M, int K, int P) {
  __shared__ __attribute__((aligned(64))) _Float16 ldsA[128][32];  // [m][k]
  __shared__ __attribute__((aligned(64))) _Float16 ldsB[128][32];  // [n][k]

  const int bb   = blockIdx.z;
  const int m0   = blockIdx.y * 128;
  const int p0   = blockIdx.x * 128;
  const int tid  = threadIdx.x;
  const int lane = tid & 31;
  const int wave = tid >> 5;
  const int mw   = m0 + (wave & 3) * 32;   // wave m base (32 rows)
  const int nh   = (wave >> 2) * 64;       // wave n base within block tile

  const _Float16* Xb = X + (size_t)bb * K * P;

  v8f acc[2][4] = {};

  for (int k0 = 0; k0 < K; k0 += 32) {
    // Stage A: 128 rows x 32 K, vectorized 16B chunks
    for (int i = tid; i < 512; i += 256) {
      int r = i >> 2, cc = (i & 3) << 3;
      *(v8h*)&ldsA[r][cc] = *(const v8h*)&Wt[(size_t)(m0 + r) * K + k0 + cc];
    }
    // Stage B transposed [n][k]: one 16B global load -> 8 element LDS stores
    for (int i = tid; i < 512; i += 256) {
      int k  = i >> 4;
      int n0 = (i & 15) << 3;
      v8h g = *(const v8h*)&Xb[(size_t)(k0 + k) * P + p0 + n0];
#pragma unroll
      for (int j = 0; j < 8; ++j) ldsB[n0 + j][k] = g[j];
    }
    if (k0 + 32 < K)
      __builtin_prefetch(&Xb[(size_t)(k0 + 32) * P + p0 + (tid & 127)], 0, 1);
    __syncthreads();

    const int ca = (lane >> 4) << 4;
    const int cb = (lane >> 4) << 4;
#pragma unroll
    for (int mi = 0; mi < 2; ++mi) {
      const int mA = mw - m0 + (mi << 4) + (lane & 15);
      v16h a = pack16(*(const v8h*)&ldsA[mA][ca],
                      *(const v8h*)&ldsA[mA][ca + 8]);
#pragma unroll
      for (int nt = 0; nt < 4; ++nt) {
        v16h bf = *(const v16h*)&ldsB[nh + (nt << 4) + (lane & 15)][cb];
        acc[mi][nt] = wmma_f16(a, bf, acc[mi][nt]);
      }
    }
    __syncthreads();
  }

#pragma unroll
  for (int mi = 0; mi < 2; ++mi) {
    float sc[8], sh[8];
#pragma unroll
    for (int g = 0; g < 8; ++g) {
      int m = mw + (mi << 4) + ((lane >> 4) << 3) + g;
      if (FUSE_BN) {
        float s = bn_g[m] * rsqrtf(bn_v[m] + 1e-5f);
        sc[g] = s;
        sh[g] = bn_b[m] - bn_m[m] * s;
      } else {
        sc[g] = 1.0f;
        sh[g] = 0.0f;
      }
    }
#pragma unroll
    for (int nt = 0; nt < 4; ++nt) {
#pragma unroll
      for (int g = 0; g < 8; ++g) {
        int m = mw + (mi << 4) + ((lane >> 4) << 3) + g;
        int p = p0 + nh + (nt << 4) + (lane & 15);
        float val = acc[mi][nt][g] * sc[g] + sh[g];
        size_t oi = (size_t)bb * M * P + (size_t)m * P + p;
        if (OUT_F16) {
          OutH[oi] = (_Float16)val;
        } else {
          if (ACCUM) val += OutF[oi];
          OutF[oi] = val;
        }
      }
    }
  }
}

// ---------------------------------------------------------------------------
// 3x3 conv, implicit GEMM (K = c*9 + tap = 2304), BN folded, accumulates.
// n-tile = 128 = one image row; tap decode hoisted per 8-element chunk.
// ---------------------------------------------------------------------------
__global__ __launch_bounds__(256) void conv3x3_wmma_kernel(
    const _Float16* __restrict__ Wt,   // [256][2304] f16
    const _Float16* __restrict__ X,    // [B][256][128][128] f16
    float* __restrict__ OutF,
    const float* __restrict__ bn_g, const float* __restrict__ bn_b,
    const float* __restrict__ bn_m, const float* __restrict__ bn_v) {
  __shared__ __attribute__((aligned(64))) _Float16 ldsA[128][32];
  __shared__ __attribute__((aligned(64))) _Float16 ldsB[128][32];

  const int bb   = blockIdx.z;
  const int m0   = blockIdx.y * 128;
  const int row  = blockIdx.x;          // one 128-wide row per block
  const int tid  = threadIdx.x;
  const int lane = tid & 31;
  const int wave = tid >> 5;
  const int mw   = m0 + (wave & 3) * 32;
  const int nh   = (wave >> 2) * 64;

  const _Float16* Xb = X + (size_t)bb * CDIM * CP;

  v8f acc[2][4] = {};

  for (int k0 = 0; k0 < 2304; k0 += 32) {
    for (int i = tid; i < 512; i += 256) {
      int r = i >> 2, cc = (i & 3) << 3;
      *(v8h*)&ldsA[r][cc] =
          *(const v8h*)&Wt[(size_t)(m0 + r) * 2304 + k0 + cc];
    }
    for (int i = tid; i < 512; i += 256) {
      int k  = i >> 4;
      int n0 = (i & 15) << 3;
      int kk = k0 + k;
      int ci = kk / 9;
      int t  = kk - ci * 9;
      int dy = t / 3 - 1;
      int dx = t - (t / 3) * 3 - 1;
      int r2 = row + dy;
      _Float16 tmp[8];
      if ((unsigned)r2 < 128u) {
        int c0 = n0 + dx;
        const _Float16* src = &Xb[(size_t)ci * CP + r2 * CW];
        if (c0 >= 0 && c0 + 7 < 128) {
          __builtin_memcpy(tmp, src + c0, 16);  // unaligned 16B load
        } else {
#pragma unroll
          for (int j = 0; j < 8; ++j) {
            int c2 = c0 + j;
            tmp[j] = ((unsigned)c2 < 128u) ? src[c2] : (_Float16)0.0f;
          }
        }
      } else {
#pragma unroll
        for (int j = 0; j < 8; ++j) tmp[j] = (_Float16)0.0f;
      }
#pragma unroll
      for (int j = 0; j < 8; ++j) ldsB[n0 + j][k] = tmp[j];
    }
    __syncthreads();

    const int ca = (lane >> 4) << 4;
    const int cb = (lane >> 4) << 4;
#pragma unroll
    for (int mi = 0; mi < 2; ++mi) {
      const int mA = mw - m0 + (mi << 4) + (lane & 15);
      v16h a = pack16(*(const v8h*)&ldsA[mA][ca],
                      *(const v8h*)&ldsA[mA][ca + 8]);
#pragma unroll
      for (int nt = 0; nt < 4; ++nt) {
        v16h bf = *(const v16h*)&ldsB[nh + (nt << 4) + (lane & 15)][cb];
        acc[mi][nt] = wmma_f16(a, bf, acc[mi][nt]);
      }
    }
    __syncthreads();
  }

#pragma unroll
  for (int mi = 0; mi < 2; ++mi) {
    float sc[8], sh[8];
#pragma unroll
    for (int g = 0; g < 8; ++g) {
      int m = mw + (mi << 4) + ((lane >> 4) << 3) + g;
      float s = bn_g[m] * rsqrtf(bn_v[m] + 1e-5f);
      sc[g] = s;
      sh[g] = bn_b[m] - bn_m[m] * s;
    }
#pragma unroll
    for (int nt = 0; nt < 4; ++nt) {
#pragma unroll
      for (int g = 0; g < 8; ++g) {
        int m = mw + (mi << 4) + ((lane >> 4) << 3) + g;
        int p = row * CW + nh + (nt << 4) + (lane & 15);
        float val = acc[mi][nt][g] * sc[g] + sh[g];
        size_t oi = (size_t)bb * CDIM * CP + (size_t)m * CP + p;
        OutF[oi] = OutF[oi] + val;
      }
    }
  }
}

// ---------------------------------------------------------------------------
// Windowed attention: one wave per (window, head).
// ---------------------------------------------------------------------------
__global__ __launch_bounds__(32) void attn_kernel(
    const _Float16* __restrict__ qkv,       // [B][768][128][128] f16
    const float* __restrict__ bias_table,   // [225][16]
    const int* __restrict__ rel_index,      // [4096]
    float* __restrict__ o) {                // [B][256][128][128]
  __shared__ __attribute__((aligned(64))) _Float16 Q[64][32];
  __shared__ __attribute__((aligned(64))) _Float16 Kt[64][32];
  __shared__ __attribute__((aligned(64))) _Float16 Vt[16][64];
  __shared__ __attribute__((aligned(64))) float    S[64][64];
  __shared__ __attribute__((aligned(64))) _Float16 Sh[64][64];

  const int lane = threadIdx.x;
  const int head = blockIdx.x & 15;
  const int win  = blockIdx.x >> 4;
  const int wx   = win & 15;
  const int wy   = (win >> 4) & 15;
  const int b    = win >> 8;
  const int h0   = wy * CWS, w0 = wx * CWS;
  const size_t base = (size_t)b * 3 * CDIM * CP;

  for (int i = lane; i < 64 * 16; i += 32) {
    int t = i >> 4, dd = i & 15;
    size_t pix = (size_t)(h0 + (t >> 3)) * CW + (w0 + (t & 7));
    int ch = head * 16 + dd;
    Q[t][dd]       = qkv[base + (size_t)(ch) * CP + pix];
    Kt[t][dd]      = qkv[base + (size_t)(CDIM + ch) * CP + pix];
    Vt[dd][t]      = qkv[base + (size_t)(2 * CDIM + ch) * CP + pix];
    Q[t][16 + dd]  = (_Float16)0.0f;
    Kt[t][16 + dd] = (_Float16)0.0f;
  }
  __syncthreads();

#pragma unroll
  for (int mt = 0; mt < 4; ++mt) {
    const int mA = (mt << 4) + (lane & 15);
    const int ca = (lane >> 4) << 4;
    v16h a = pack16(*(const v8h*)&Q[mA][ca], *(const v8h*)&Q[mA][ca + 8]);
#pragma unroll
    for (int nt = 0; nt < 4; ++nt) {
      const int nB = (nt << 4) + (lane & 15);
      const int cb = (lane >> 4) << 4;
      v16h bf = *(const v16h*)&Kt[nB][cb];
      v8f c = {};
      c = wmma_f16(a, bf, c);
#pragma unroll
      for (int g = 0; g < 8; ++g)
        S[(mt << 4) + ((lane >> 4) << 3) + g][(nt << 4) + (lane & 15)] = c[g];
    }
  }
  __syncthreads();

  float* Sf = &S[0][0];
  for (int i = lane; i < 4096; i += 32)
    Sf[i] = Sf[i] * 0.25f + bias_table[rel_index[i] * CHEADS + head];
  __syncthreads();

  for (int rr = 0; rr < 2; ++rr) {
    int r = lane + rr * 32;
    float mx = -1e30f;
    for (int j = 0; j < 64; ++j) mx = fmaxf(mx, S[r][j]);
    float sum = 0.0f;
    for (int j = 0; j < 64; ++j) {
      float e = __expf(S[r][j] - mx);
      S[r][j] = e;
      sum += e;
    }
    float inv = 1.0f / sum;
    for (int j = 0; j < 64; ++j) Sh[r][j] = (_Float16)(S[r][j] * inv);
  }
  __syncthreads();

#pragma unroll
  for (int mt = 0; mt < 4; ++mt) {
    v8f acc = {};
#pragma unroll
    for (int ks = 0; ks < 2; ++ks) {
      const int mA = (mt << 4) + (lane & 15);
      const int ca = ks * 32 + ((lane >> 4) << 4);
      v16h a = pack16(*(const v8h*)&Sh[mA][ca], *(const v8h*)&Sh[mA][ca + 8]);
      const int cb = ks * 32 + ((lane >> 4) << 4);
      v16h bf = *(const v16h*)&Vt[lane & 15][cb];
      acc = wmma_f16(a, bf, acc);
    }
    const int nn = lane & 15;
    const size_t chb = (size_t)b * CDIM + head * 16 + nn;
#pragma unroll
    for (int g = 0; g < 8; ++g) {
      int m = (mt << 4) + ((lane >> 4) << 3) + g;
      int hh = h0 + (m >> 3), ww = w0 + (m & 7);
      o[chb * CP + hh * CW + ww] = acc[g];
    }
  }
}

// ---------------------------------------------------------------------------
__global__ void pool_fuse_kernel(const float* __restrict__ o,
                                 const float* __restrict__ local,
                                 float* __restrict__ pre) {
  int idx = blockIdx.x * blockDim.x + threadIdx.x;
  if (idx >= CB * CDIM * CP) return;
  int w = idx & 127;
  int h = (idx >> 7) & 127;
  int bc = idx >> 14;
  const float* op = o + (size_t)bc * CP;
  float ax = 0.0f, ay = 0.0f;
#pragma unroll
  for (int j = 0; j < 8; ++j) {
    int r = h - 3 + j;
    if (r >= 0 && r < 129) {
      int rr = (r == 128) ? 126 : r;
      ax += op[rr * CW + w];
    }
    int c = w - 3 + j;
    if (c >= 0 && c < 129) {
      int cc = (c == 128) ? 126 : c;
      ay += op[h * CW + cc];
    }
  }
  pre[idx] = ax * 0.125f + ay * 0.125f + local[idx];
}

// ---------------------------------------------------------------------------
__global__ void dw_bn_kernel(const float* __restrict__ pre,
                             const float* __restrict__ wdw,  // [256][64] f32
                             const float* __restrict__ g, const float* __restrict__ b,
                             const float* __restrict__ m, const float* __restrict__ v,
                             _Float16* __restrict__ outh) {
  int idx = blockIdx.x * blockDim.x + threadIdx.x;
  if (idx >= CB * CDIM * CP) return;
  int w = idx & 127;
  int h = (idx >> 7) & 127;
  int bc = idx >> 14;
  int c = bc & 255;
  const float* pp = pre + (size_t)bc * CP;
  const float* wc = wdw + c * 64;
  float acc = 0.0f;
#pragma unroll
  for (int ky = 0; ky < 8; ++ky) {
    int r = h - 3 + ky;
    if (r < 0 || r >= 129) continue;
    int rr = (r == 128) ? 126 : r;
#pragma unroll
    for (int kx = 0; kx < 8; ++kx) {
      int col = w - 3 + kx;
      if (col < 0 || col >= 129) continue;
      int cc = (col == 128) ? 126 : col;
      acc += pp[rr * CW + cc] * wc[ky * 8 + kx];
    }
  }
  float sc = g[c] * rsqrtf(v[c] + 1e-5f);
  outh[idx] = (_Float16)(acc * sc + (b[c] - m[c] * sc));
}

// ---------------------------------------------------------------------------
extern "C" void kernel_launch(void* const* d_in, const int* in_sizes, int n_in,
                              void* d_out, int out_size, void* d_ws, size_t ws_size,
                              hipStream_t stream) {
  const float* x        = (const float*)d_in[0];
  const float* y        = (const float*)d_in[1];
  const float* qkv_w    = (const float*)d_in[2];
  const float* local1_w = (const float*)d_in[3];
  const float* l1_g = (const float*)d_in[4];
  const float* l1_b = (const float*)d_in[5];
  const float* l1_m = (const float*)d_in[6];
  const float* l1_v = (const float*)d_in[7];
  const float* local2_w = (const float*)d_in[8];
  const float* l2_g = (const float*)d_in[9];
  const float* l2_b = (const float*)d_in[10];
  const float* l2_m = (const float*)d_in[11];
  const float* l2_v = (const float*)d_in[12];
  const float* proj_dw_w = (const float*)d_in[13];
  const float* p_g = (const float*)d_in[14];
  const float* p_b = (const float*)d_in[15];
  const float* p_m = (const float*)d_in[16];
  const float* p_v = (const float*)d_in[17];
  const float* proj_pw_w = (const float*)d_in[18];
  const float* relb = (const float*)d_in[19];
  const int*   reli = (const int*)d_in[20];
  float* out = (float*)d_out;
  (void)in_sizes; (void)n_in; (void)out_size; (void)ws_size;

  const size_t NX = (size_t)CB * CDIM * CP;

  char* base = (char*)d_ws;
  size_t off = 0;
  auto alloc = [&](size_t bytes) {
    size_t o = off;
    off += (bytes + 255) & ~(size_t)255;
    return o;
  };
  size_t off_hx   = alloc(NX * 2);
  size_t off_hy   = alloc(NX * 2);
  size_t off_qkvw = alloc((size_t)768 * 256 * 2);
  size_t off_l1w  = alloc((size_t)256 * 2304 * 2);
  size_t off_l2w  = alloc((size_t)256 * 256 * 2);
  size_t off_pww  = alloc((size_t)256 * 256 * 2);
  size_t off_hqkv = alloc((size_t)CB * 768 * CP * 2);
  size_t off_loc  = alloc(NX * 4);
  size_t off_o    = alloc(NX * 4);
  size_t off_pre  = off_hx;    // alias: x/y f16 regions dead by pooling
  size_t off_dwbn = off_hqkv;  // alias: qkv region dead after attention

  _Float16* h_x   = (_Float16*)(base + off_hx);
  _Float16* h_y   = (_Float16*)(base + off_hy);
  _Float16* h_qw  = (_Float16*)(base + off_qkvw);
  _Float16* h_l1w = (_Float16*)(base + off_l1w);
  _Float16* h_l2w = (_Float16*)(base + off_l2w);
  _Float16* h_pww = (_Float16*)(base + off_pww);
  _Float16* h_qkv = (_Float16*)(base + off_hqkv);
  float*    f_loc = (float*)(base + off_loc);
  float*    f_o   = (float*)(base + off_o);
  float*    f_pre = (float*)(base + off_pre);
  _Float16* h_dw  = (_Float16*)(base + off_dwbn);

  int nb = (int)((NX + 255) / 256);
  cvt_f16_kernel<<<nb, 256, 0, stream>>>(x, h_x, (int)NX);
  cvt_f16_kernel<<<nb, 256, 0, stream>>>(y, h_y, (int)NX);
  cvt_f16_kernel<<<(768 * 256 + 255) / 256, 256, 0, stream>>>(qkv_w, h_qw, 768 * 256);
  cvt_f16_kernel<<<(256 * 2304 + 255) / 256, 256, 0, stream>>>(local1_w, h_l1w, 256 * 2304);
  cvt_f16_kernel<<<(256 * 256 + 255) / 256, 256, 0, stream>>>(local2_w, h_l2w, 256 * 256);
  cvt_f16_kernel<<<(256 * 256 + 255) / 256, 256, 0, stream>>>(proj_pw_w, h_pww, 256 * 256);

  // qkv = conv1x1(x): M=768 K=256 -> f16
  wmma_gemm_kernel<0, 0, 1><<<dim3(CP / 128, 768 / 128, CB), 256, 0, stream>>>(
      h_qw, h_x, nullptr, h_qkv, nullptr, nullptr, nullptr, nullptr,
      768, 256, CP);

  // local = BN(conv1x1(y))
  wmma_gemm_kernel<1, 0, 0><<<dim3(CP / 128, 256 / 128, CB), 256, 0, stream>>>(
      h_l2w, h_y, f_loc, nullptr, l2_g, l2_b, l2_m, l2_v, 256, 256, CP);

  // local += BN(conv3x3(y))
  conv3x3_wmma_kernel<<<dim3(CH, 256 / 128, CB), 256, 0, stream>>>(
      h_l1w, h_y, f_loc, l1_g, l1_b, l1_m, l1_v);

  // windowed attention
  attn_kernel<<<CB * 16 * 16 * CHEADS, 32, 0, stream>>>(h_qkv, relb, reli, f_o);

  // pools + fuse
  pool_fuse_kernel<<<nb, 256, 0, stream>>>(f_o, f_loc, f_pre);

  // depthwise 8x8 + BN -> f16
  dw_bn_kernel<<<nb, 256, 0, stream>>>(f_pre, proj_dw_w, p_g, p_b, p_m, p_v, h_dw);

  // projection 1x1 -> d_out
  wmma_gemm_kernel<0, 0, 0><<<dim3(CP / 128, 256 / 128, CB), 256, 0, stream>>>(
      h_pww, h_dw, out, nullptr, nullptr, nullptr, nullptr, nullptr,
      256, 256, CP);
}
